// PatchClusterVAE_26886495273682
// MI455X (gfx1250) — matrix operand order repaired and checked by reading, not compiled, not run on hardware
//
#include <hip/hip_runtime.h>

#define B_   32
#define L_   2048
#define C_   32
#define PL_  16
#define P_   128
#define HE_  256
#define HD_  256
#define K_   512
#define NH_  4
#define EPS_ 1e-5f
#define BETA_ 0.25f
#define CW_  0.2f

typedef __attribute__((ext_vector_type(16))) _Float16 v16h;
typedef __attribute__((ext_vector_type(8)))  float    v8f;

#if defined(__gfx1250__) && __has_builtin(__builtin_amdgcn_tensor_load_to_lds) && \
    __has_builtin(__builtin_amdgcn_s_wait_tensorcnt)
#define HAVE_TDM 1
#else
#define HAVE_TDM 0
#endif

#if HAVE_TDM
typedef unsigned int uv4 __attribute__((ext_vector_type(4)));
typedef int          iv4 __attribute__((ext_vector_type(4)));
typedef int          iv8 __attribute__((ext_vector_type(8)));

// TDM: DMA one strided f32 column x[b, 0:2048, c] (stride C*4B) into contiguous LDS.
// D# group0: count=1 | lds_addr | global_addr(57b) | type=2
// D# group1: data_size=4B, tile = 1 x 2048, tensor_dim0_stride = C
__device__ __forceinline__ void tdm_load_xcol(void* lds_dst, const float* gsrc) {
  unsigned long long ga = (unsigned long long)(uintptr_t)gsrc;
  uv4 g0;
  g0[0] = 1u;                                             // count=1 (valid descriptor)
  g0[1] = (unsigned)((uintptr_t)lds_dst & 0xffffffffu);   // lds_addr
  g0[2] = (unsigned)(ga & 0xffffffffu);                   // global_addr[31:0]
  g0[3] = (unsigned)((ga >> 32) & 0x01ffffffu) | (2u << 30);  // addr[56:32] | type=2
  iv8 g1;
  g1[0] = 0x00020000;            // workgroup_mask=0, data_size=2 (4 bytes)
  g1[1] = (int)(32u << 16);      // tensor_dim0 = 32 (C)
  g1[2] = (int)(2048u << 16);    // tensor_dim1 = 2048 (L)
  g1[3] = (int)(1u << 16);       // tile_dim0 = 1
  g1[4] = 2048;                  // tile_dim1 = 2048
  g1[5] = 32;                    // tensor_dim0_stride = C
  g1[6] = 0;
  g1[7] = 0;
  iv4 z4 = {0, 0, 0, 0};
#if __clang_major__ >= 23
  iv8 z8 = {0, 0, 0, 0, 0, 0, 0, 0};
  __builtin_amdgcn_tensor_load_to_lds(g0, g1, z4, z4, z8, 0);
#else
  __builtin_amdgcn_tensor_load_to_lds(g0, g1, z4, z4, 0);
#endif
}
#endif  // HAVE_TDM

// ---------------- WMMA fragment helpers (gfx1250, wave32) ----------------

__device__ __forceinline__ v8f wmma16(v16h a, v16h b, v8f c) {
  return __builtin_amdgcn_wmma_f32_16x16x32_f16(false, a, false, b, (short)0, c,
                                                false, false);
}

// A fragment, 16-bit 16x32: src row-major [M][ld] (halves), K window [k0,k0+kvalid)
// lanes 0-15: V0-7 => K0..15 ; lanes 16-31: K16..31. Pairs are contiguous
// dwords (ko even, kvalid even => pair all-valid or all-zero).
__device__ __forceinline__ v16h load_fragA(const _Float16* src, int ld, int row0,
                                           int k0, int kvalid) {
  int lane = threadIdx.x & 31;
  const _Float16* rp = src + (row0 + (lane & 15)) * ld + k0;
  int kb = (lane >> 4) << 4;
  union { v16h h; unsigned u[8]; } f;
#pragma unroll
  for (int j = 0; j < 8; ++j) {
    int ko = kb + 2 * j;
    f.u[j] = (ko < kvalid) ? *(const unsigned*)(rp + ko) : 0u;
  }
  return f.h;
}

// B fragment from pre-packed storage: tile (kt,nt) stored as [lane][8] dwords.
__device__ __forceinline__ v16h load_fragB_packed(const unsigned* base, int ntiles,
                                                  int kt, int nt) {
  int lane = threadIdx.x & 31;
  const unsigned* p = base + ((kt * ntiles + nt) * 32 + lane) * 8;
  union { v16h h; unsigned u[8]; } f;
#pragma unroll
  for (int j = 0; j < 8; ++j) f.u[j] = p[j];
  return f.h;
}

// B fragment from transposed source: B[k][n] = src[n*ld + k] (src row-major [N][ld])
// lane n = l%16 ; K base = (l/16)*16 ; pairs contiguous dwords.
__device__ __forceinline__ v16h load_fragBT(const _Float16* src, int ld, int k0,
                                            int n0, int kvalid) {
  int lane = threadIdx.x & 31;
  const _Float16* rp = src + (n0 + (lane & 15)) * ld + k0;
  int kb = (lane >> 4) << 4;
  union { v16h h; unsigned u[8]; } f;
#pragma unroll
  for (int j = 0; j < 8; ++j) {
    int ko = kb + 2 * j;
    f.u[j] = (ko < kvalid) ? *(const unsigned*)(rp + ko) : 0u;
  }
  return f.h;
}

__device__ __forceinline__ unsigned pack2h(float lo, float hi) {
  unsigned short a = __builtin_bit_cast(unsigned short, (_Float16)lo);
  unsigned short b = __builtin_bit_cast(unsigned short, (_Float16)hi);
  return (unsigned)a | ((unsigned)b << 16);
}

// ---------------- workspace layout ----------------
// bytes [0,4096): accumulators (0: rec_sum, 1: cluster_sum, 2..513: counts[512])
// bytes [4096,8192): mean[1024] ; [8192,12288): std[1024]
// bytes [12288,...): packed f16 weights (dword units below) ; [223232,...): e2[512]
#define DOFF_INPW 0       // K32(valid16) x N64   : 1024 dwords
#define DOFF_WQ   1024    // 64x64                : 2048
#define DOFF_WK   3072
#define DOFF_WV   5120
#define DOFF_WO   7168
#define DOFF_FW1  9216    // 64x256               : 8192
#define DOFF_FW2  17408   // 256x64               : 8192
#define DOFF_DW1  25600   // 64x256               : 8192
#define DOFF_DW2  33792   // 256x16               : 2048
#define DOFF_DWR  35840   // 64x16                : 512
#define DOFF_CBT  36352   // codebook^T 64x512    : 16384 (end 52736)

// ---------------- kernel 0: weight prep (pack into B-fragment order) ----------------

struct PrepArgs {
  const float *inp_W, *Wq, *Wk, *Wv, *Wo, *fW1, *fW2, *cb, *dW1, *dW2, *dWr;
  float* ws_acc;
  float* ws_e2;
  unsigned* wd;
};

__global__ __launch_bounds__(256) void prep_kernel(PrepArgs a) {
  int tid = blockIdx.x * blockDim.x + threadIdx.x;
  int nT  = gridDim.x * blockDim.x;
  for (int i = tid; i < 514; i += nT) a.ws_acc[i] = 0.f;
  // inp_W: K valid 16, padded to 32, N=64 (1 ktile x 4 ntiles)
  for (int i = tid; i < 1024; i += nT) {
    int tile = i >> 8, r = i & 255, lane = r >> 3, j = r & 7;
    int nn = tile * 16 + (lane & 15);
    int k  = ((lane >> 4) << 4) + 2 * j;
    float lo = (k < 16)     ? a.inp_W[k * 64 + nn]       : 0.f;
    float hi = (k + 1 < 16) ? a.inp_W[(k + 1) * 64 + nn] : 0.f;
    a.wd[DOFF_INPW + i] = pack2h(lo, hi);
  }
  const float* srcs[9] = {a.Wq, a.Wk, a.Wv, a.Wo, a.fW1, a.fW2, a.dW1, a.dW2, a.dWr};
  const int    offs[9] = {DOFF_WQ, DOFF_WK, DOFF_WV, DOFF_WO, DOFF_FW1,
                          DOFF_FW2, DOFF_DW1, DOFF_DW2, DOFF_DWR};
  const int    Ks[9]   = {64, 64, 64, 64, 64, 256, 64, 256, 64};
  const int    Ns[9]   = {64, 64, 64, 64, 256, 64, 256, 16, 16};
  for (int m = 0; m < 9; ++m) {
    int K = Ks[m], N = Ns[m], ntl = N >> 4;
    int nd = (K * N) >> 1;
    const float* s = srcs[m];
    for (int i = tid; i < nd; i += nT) {
      int tile = i >> 8, r = i & 255, lane = r >> 3, j = r & 7;
      int kt = tile / ntl, nt = tile % ntl;
      int nn = nt * 16 + (lane & 15);
      int k  = kt * 32 + ((lane >> 4) << 4) + 2 * j;
      a.wd[offs[m] + i] = pack2h(s[k * N + nn], s[(k + 1) * N + nn]);
    }
  }
  // codebook^T: B[k][n] = cb[n*64 + k], K=64, N=512 (2 kt x 32 nt)
  for (int i = tid; i < 16384; i += nT) {
    int tile = i >> 8, r = i & 255, lane = r >> 3, j = r & 7;
    int kt = tile >> 5, nt = tile & 31;
    int nn = nt * 16 + (lane & 15);
    int k  = kt * 32 + ((lane >> 4) << 4) + 2 * j;
    a.wd[DOFF_CBT + i] = pack2h(a.cb[nn * 64 + k], a.cb[nn * 64 + k + 1]);
  }
  for (int k = tid; k < 512; k += nT) {
    const float* r = a.cb + k * 64;
    float s = 0.f;
    for (int j = 0; j < 64; ++j) s += r[j] * r[j];
    a.ws_e2[k] = s;
  }
}

// ---------------- kernel 1: RevIN stats (TDM column DMA) ----------------

__global__ __launch_bounds__(256) void revin_kernel(const float* __restrict__ x,
                                                    float* ws_mean, float* ws_std) {
  __shared__ float xs[L_];
  __shared__ float s1[256], s2[256];
  int nn = blockIdx.x;
  int b = nn >> 5, c = nn & 31;
  float s = 0.f, sq = 0.f;
#if HAVE_TDM
  if (threadIdx.x == 0) {
    tdm_load_xcol(xs, x + (size_t)b * (L_ * C_) + c);
    __builtin_amdgcn_s_wait_tensorcnt(0);
  }
  __syncthreads();
  for (int l = threadIdx.x; l < L_; l += 256) {
    float v = xs[l];
    s += v; sq += v * v;
  }
#else
  for (int l = threadIdx.x; l < L_; l += 256) {
    float v = x[(b * L_ + l) * C_ + c];
    s += v; sq += v * v;
  }
#endif
  s1[threadIdx.x] = s; s2[threadIdx.x] = sq;
  __syncthreads();
  for (int st = 128; st > 0; st >>= 1) {
    if (threadIdx.x < st) {
      s1[threadIdx.x] += s1[threadIdx.x + st];
      s2[threadIdx.x] += s2[threadIdx.x + st];
    }
    __syncthreads();
  }
  if (threadIdx.x == 0) {
    float mu  = s1[0] * (1.0f / L_);
    float var = s2[0] * (1.0f / L_) - mu * mu;
    ws_mean[nn] = mu;
    ws_std[nn]  = sqrtf(var + EPS_);
  }
}

// ---------------- kernel 2: fused encoder / quantizer / decoder ----------------

struct EncArgs {
  const float *x, *revin_w, *revin_b;
  const float *inp_b, *bq, *bk, *bv, *bo;
  const float *ln1_g, *ln1_b, *ffn_b1, *ffn_b2, *ln2_g, *ln2_b;
  const float *codebook, *dec_b1, *dec_b2, *dec_br, *dec_ln_g, *dec_ln_b;
  const float *ws_mean, *ws_std, *ws_e2;
  const unsigned* wd;   // packed f16 weights (dword view)
  float* ws_acc;
  float *out_zq, *out_r, *out_idx;
};

#define SMEM_X16   32768
#define SMEM_S     49152
#define SMEM_MISC  147456
#define SMEM_TOTAL 151552

__global__ __launch_bounds__(256, 1) void enc_kernel(EncArgs A) {
  __shared__ __align__(16) char smem[SMEM_TOTAL];
  float*    h32 = (float*)smem;                     // [128][64] residual stream f32
  _Float16* x16 = (_Float16*)(smem + SMEM_X16);     // [128][64] activation f16
  char*     S   = smem + SMEM_S;                    // 96KB stage-aliased scratch
  float*    red   = (float*)(smem + SMEM_MISC);     // [256]
  float*    bestd = red + 256;                      // [128]
  int*      besti = (int*)(bestd + 128);            // [128]
  float*    z2s   = (float*)(besti + 128);          // [128]

  const int tid  = threadIdx.x;
  const int w    = tid >> 5;
  const int lane = tid & 31;
  const int m0   = w * 16;
  const int n    = blockIdx.x;
  const int b    = n >> 5, c = n & 31;

  __builtin_prefetch((const char*)A.wd + tid * 824, 0, 1);  // warm packed weights

  const float mean = A.ws_mean[n];
  const float stdv = A.ws_std[n];
  const float rstd = 1.0f / stdv;
  const float rw   = A.revin_w[c], rb = A.revin_b[c];
  const unsigned* wd = A.wd;

  // ---- Stage A: DMA column, RevIN-normalize, patchify (K 16 -> padded 32) ----
  _Float16* pat16 = (_Float16*)S;  // [128][32]
#if HAVE_TDM
  float* xs = (float*)(S + 8192);  // [2048] staging
  if (tid == 0) {
    tdm_load_xcol(xs, A.x + (size_t)b * (L_ * C_) + c);
    __builtin_amdgcn_s_wait_tensorcnt(0);
  }
  __syncthreads();
  for (int i = tid; i < L_; i += 256) {
    float xn = (xs[i] - mean) * rstd * rw + rb;
    pat16[(i >> 4) * 32 + (i & 15)]      = (_Float16)xn;
    pat16[(i >> 4) * 32 + 16 + (i & 15)] = (_Float16)0.f;
  }
#else
  for (int i = tid; i < L_; i += 256) {
    float xv = A.x[(b * L_ + i) * C_ + c];
    float xn = (xv - mean) * rstd * rw + rb;
    pat16[(i >> 4) * 32 + (i & 15)]      = (_Float16)xn;
    pat16[(i >> 4) * 32 + 16 + (i & 15)] = (_Float16)0.f;
  }
#endif
  __syncthreads();

  // h = pat @ inp_W + inp_b -> h32 + x16
  {
    v16h a = load_fragA(pat16, 32, m0, 0, 32);
#pragma unroll
    for (int nt = 0; nt < 4; ++nt) {
      v16h bf = load_fragB_packed(wd + DOFF_INPW, 4, 0, nt);
      v8f acc = {};
      acc = wmma16(a, bf, acc);
      int col = nt * 16 + (lane & 15);
      float bias = A.inp_b[col];
#pragma unroll
      for (int r = 0; r < 8; ++r) {
        int row = m0 + r + ((lane >> 4) << 3);
        float v = acc[r] + bias;
        h32[row * 64 + col] = v;
        x16[row * 64 + col] = (_Float16)v;
      }
    }
  }
  __syncthreads();

  // ---- Stage B: QKV (V written in packed-B layout for att@V) ----
  _Float16* q16  = (_Float16*)S;              // [128][64]
  _Float16* k16  = (_Float16*)(S + 16384);    // [128][64]
  _Float16* vpk  = (_Float16*)(S + 32768);    // packed-B tiles of V (16KB)
  _Float16* sc16 = (_Float16*)(S + 49152);    // [128][128]
  _Float16* a16  = (_Float16*)(S + 81920);    // [128][64]
  {
    const int     woff[3] = {DOFF_WQ, DOFF_WK, DOFF_WV};
    const float*  bs[3]   = {A.bq, A.bk, A.bv};
    for (int m = 0; m < 3; ++m) {
      for (int nt = 0; nt < 4; ++nt) {
        v8f acc = {};
#pragma unroll
        for (int kt = 0; kt < 2; ++kt) {
          v16h a  = load_fragA(x16, 64, m0, kt * 32, 32);
          v16h bf = load_fragB_packed(wd + woff[m], 4, kt, nt);
          acc = wmma16(a, bf, acc);
        }
        int col = nt * 16 + (lane & 15);
        float bias = bs[m][col];
#pragma unroll
        for (int r = 0; r < 8; ++r) {
          int row = m0 + r + ((lane >> 4) << 3);
          _Float16 val = (_Float16)(acc[r] + bias);
          if (m == 0) {
            q16[row * 64 + col] = val;
          } else if (m == 1) {
            k16[row * 64 + col] = val;
          } else {  // V -> packed-B tile layout: tile (row/32, nt)
            int kk  = row & 31;
            int idx = (((row >> 5) * 4 + nt) * 32 +
                       ((col & 15) | (((kk >> 4) & 1) << 4))) * 8 + ((kk & 15) >> 1);
            vpk[idx * 2 + (kk & 1)] = val;
          }
        }
      }
    }
  }
  __syncthreads();

  // ---- Stage C: attention per head (dh=16 padded to K=32) ----
  for (int hd = 0; hd < NH_; ++hd) {
    {
      v16h a = load_fragA(q16, 64, m0, hd * 16, 16);
      for (int nt = 0; nt < 8; ++nt) {
        v16h bf = load_fragBT(k16, 64, hd * 16, nt * 16, 16);  // k^T
        v8f acc = {};
        acc = wmma16(a, bf, acc);
        int col = nt * 16 + (lane & 15);
#pragma unroll
        for (int r = 0; r < 8; ++r) {
          int row = m0 + r + ((lane >> 4) << 3);
          sc16[row * 128 + col] = (_Float16)(acc[r] * 0.25f);  // 1/sqrt(dh)
        }
      }
    }
    __syncthreads();
    if (tid < 128) {  // softmax over k per row
      _Float16* rowp = sc16 + tid * 128;
      float mx = -1e30f;
      for (int j = 0; j < 128; ++j) mx = fmaxf(mx, (float)rowp[j]);
      float sm = 0.f;
      for (int j = 0; j < 128; ++j) {
        float e = __expf((float)rowp[j] - mx);
        sm += e;
        rowp[j] = (_Float16)e;
      }
      float inv = 1.0f / sm;
      for (int j = 0; j < 128; ++j) rowp[j] = (_Float16)((float)rowp[j] * inv);
    }
    __syncthreads();
    {  // a_h = att @ v_h  (V already in packed-B tiles, ntiles=4, nt=hd)
      v8f acc = {};
#pragma unroll
      for (int kt = 0; kt < 4; ++kt) {
        v16h a  = load_fragA(sc16, 128, m0, kt * 32, 32);
        v16h bf = load_fragB_packed((const unsigned*)vpk, 4, kt, hd);
        acc = wmma16(a, bf, acc);
      }
      int col = hd * 16 + (lane & 15);
#pragma unroll
      for (int r = 0; r < 8; ++r) {
        int row = m0 + r + ((lane >> 4) << 3);
        a16[row * 64 + col] = (_Float16)acc[r];
      }
    }
    __syncthreads();
  }

  // ---- Stage D: out-proj + residual + LN1 ----
  for (int nt = 0; nt < 4; ++nt) {
    v8f acc = {};
#pragma unroll
    for (int kt = 0; kt < 2; ++kt) {
      v16h a  = load_fragA(a16, 64, m0, kt * 32, 32);
      v16h bf = load_fragB_packed(wd + DOFF_WO, 4, kt, nt);
      acc = wmma16(a, bf, acc);
    }
    int col = nt * 16 + (lane & 15);
    float bias = A.bo[col];
#pragma unroll
    for (int r = 0; r < 8; ++r) {
      int row = m0 + r + ((lane >> 4) << 3);
      h32[row * 64 + col] += acc[r] + bias;
    }
  }
  __syncthreads();
  if (tid < 128) {
    float* hr = h32 + tid * 64;
    float mu = 0.f;
    for (int j = 0; j < 64; ++j) mu += hr[j];
    mu *= (1.0f / 64);
    float var = 0.f;
    for (int j = 0; j < 64; ++j) { float d = hr[j] - mu; var += d * d; }
    var *= (1.0f / 64);
    float rs = rsqrtf(var + EPS_);
    for (int j = 0; j < 64; ++j) {
      float v = (hr[j] - mu) * rs * A.ln1_g[j] + A.ln1_b[j];
      hr[j] = v;
      x16[tid * 64 + j] = (_Float16)v;
    }
  }
  __syncthreads();

  // ---- Stage E: FFN + residual + LN2 ----
  _Float16* ff16 = (_Float16*)S;  // [128][256]
  for (int nt = 0; nt < 16; ++nt) {
    v8f acc = {};
#pragma unroll
    for (int kt = 0; kt < 2; ++kt) {
      v16h a  = load_fragA(x16, 64, m0, kt * 32, 32);
      v16h bf = load_fragB_packed(wd + DOFF_FW1, 16, kt, nt);
      acc = wmma16(a, bf, acc);
    }
    int col = nt * 16 + (lane & 15);
    float bias = A.ffn_b1[col];
#pragma unroll
    for (int r = 0; r < 8; ++r) {
      int row = m0 + r + ((lane >> 4) << 3);
      ff16[row * 256 + col] = (_Float16)fmaxf(acc[r] + bias, 0.f);
    }
  }
  __syncthreads();
  for (int nt = 0; nt < 4; ++nt) {
    v8f acc = {};
#pragma unroll
    for (int kt = 0; kt < 8; ++kt) {
      v16h a  = load_fragA(ff16, 256, m0, kt * 32, 32);
      v16h bf = load_fragB_packed(wd + DOFF_FW2, 4, kt, nt);
      acc = wmma16(a, bf, acc);
    }
    int col = nt * 16 + (lane & 15);
    float bias = A.ffn_b2[col];
#pragma unroll
    for (int r = 0; r < 8; ++r) {
      int row = m0 + r + ((lane >> 4) << 3);
      h32[row * 64 + col] += acc[r] + bias;
    }
  }
  __syncthreads();
  if (tid < 128) {  // LN2 -> z (f32 in h32, f16 in x16)
    float* hr = h32 + tid * 64;
    float mu = 0.f;
    for (int j = 0; j < 64; ++j) mu += hr[j];
    mu *= (1.0f / 64);
    float var = 0.f;
    for (int j = 0; j < 64; ++j) { float d = hr[j] - mu; var += d * d; }
    var *= (1.0f / 64);
    float rs = rsqrtf(var + EPS_);
    for (int j = 0; j < 64; ++j) {
      float v = (hr[j] - mu) * rs * A.ln2_g[j] + A.ln2_b[j];
      hr[j] = v;
      x16[tid * 64 + j] = (_Float16)v;
    }
  }
  __syncthreads();

  // ---- Stage F: cluster quantizer, streamed argmin over K=512 ----
  float* ze = (float*)S;  // [128][64]
  if (tid < 128) {
    const float* hr = h32 + tid * 64;
    float s = 0.f;
    for (int j = 0; j < 64; ++j) s += hr[j] * hr[j];
    z2s[tid] = s;
    bestd[tid] = 3.4e38f;
    besti[tid] = 0;
  }
  __syncthreads();
  for (int cc = 0; cc < 8; ++cc) {
    for (int nt = 0; nt < 4; ++nt) {
      v8f acc = {};
#pragma unroll
      for (int kt = 0; kt < 2; ++kt) {
        v16h a  = load_fragA(x16, 64, m0, kt * 32, 32);
        v16h bf = load_fragB_packed(wd + DOFF_CBT, 32, kt, cc * 4 + nt);
        acc = wmma16(a, bf, acc);
      }
      int col = nt * 16 + (lane & 15);
#pragma unroll
      for (int r = 0; r < 8; ++r) {
        int row = m0 + r + ((lane >> 4) << 3);
        ze[row * 64 + col] = acc[r];
      }
    }
    __syncthreads();
    if (tid < 128) {
      float bd = bestd[tid];
      int   bi = besti[tid];
      const float* zr = ze + tid * 64;
      float zz = z2s[tid];
      for (int j = 0; j < 64; ++j) {
        int kidx = cc * 64 + j;
        float d = zz + A.ws_e2[kidx] - 2.0f * zr[j];
        if (d < bd) { bd = d; bi = kidx; }
      }
      bestd[tid] = bd;
      besti[tid] = bi;
    }
    __syncthreads();
  }

  // emit indices, z_q_st, counts, cluster loss; x16 <- z_q (f16)
  float clup = 0.f;
  if (tid < 128) {
    int bi = besti[tid];
    A.out_idx[n * P_ + tid] = (float)bi;
    atomicAdd(A.ws_acc + 2 + bi, 1.0f);
    const float* cbr = A.codebook + bi * 64;
    const float* zr  = h32 + tid * 64;
    float* oz = A.out_zq + (size_t)(n * P_ + tid) * 64;
    for (int j = 0; j < 64; ++j) {
      float q = cbr[j];
      float dd = q - zr[j];
      clup += dd * dd;
      oz[j] = q;
      x16[tid * 64 + j] = (_Float16)q;
    }
  }
  red[tid] = clup;
  __syncthreads();
  for (int s = 128; s > 0; s >>= 1) {
    if (tid < s) red[tid] += red[tid + s];
    __syncthreads();
  }
  if (tid == 0) atomicAdd(A.ws_acc + 1, red[0]);
  __syncthreads();

  // ---- Stage G: decoder (overlap: kick off TDM reload of x column now) ----
  _Float16* fd16 = (_Float16*)S;            // [128][256]
  float*    yln  = (float*)(S + 65536);     // [128][16]
#if HAVE_TDM
  float*    xs2  = (float*)(S + 73728);     // [2048] staging for loss stage
  if (tid == 0) tdm_load_xcol(xs2, A.x + (size_t)b * (L_ * C_) + c);
#endif
  for (int nt = 0; nt < 16; ++nt) {
    v8f acc = {};
#pragma unroll
    for (int kt = 0; kt < 2; ++kt) {
      v16h a  = load_fragA(x16, 64, m0, kt * 32, 32);
      v16h bf = load_fragB_packed(wd + DOFF_DW1, 16, kt, nt);
      acc = wmma16(a, bf, acc);
    }
    int col = nt * 16 + (lane & 15);
    float bias = A.dec_b1[col];
#pragma unroll
    for (int r = 0; r < 8; ++r) {
      int row = m0 + r + ((lane >> 4) << 3);
      fd16[row * 256 + col] = (_Float16)fmaxf(acc[r] + bias, 0.f);
    }
  }
  __syncthreads();
  {
    v8f acc = {};
#pragma unroll
    for (int kt = 0; kt < 8; ++kt) {
      v16h a  = load_fragA(fd16, 256, m0, kt * 32, 32);
      v16h bf = load_fragB_packed(wd + DOFF_DW2, 1, kt, 0);
      acc = wmma16(a, bf, acc);
    }
#pragma unroll
    for (int kt = 0; kt < 2; ++kt) {
      v16h a  = load_fragA(x16, 64, m0, kt * 32, 32);
      v16h bf = load_fragB_packed(wd + DOFF_DWR, 1, kt, 0);
      acc = wmma16(a, bf, acc);
    }
    int col = lane & 15;
    float bias = A.dec_b2[col] + A.dec_br[col];
#pragma unroll
    for (int r = 0; r < 8; ++r) {
      int row = m0 + r + ((lane >> 4) << 3);
      yln[row * 16 + col] = acc[r] + bias;
    }
  }
#if HAVE_TDM
  if (tid == 0) __builtin_amdgcn_s_wait_tensorcnt(0);
#endif
  __syncthreads();

  // ---- Stage H: dec LN (PL) + RevIN denorm + reconstruction loss ----
  float recp = 0.f;
  if (tid < 128) {
    float* yr = yln + tid * 16;
    float mu = 0.f;
    for (int l = 0; l < 16; ++l) mu += yr[l];
    mu *= (1.0f / 16);
    float var = 0.f;
    for (int l = 0; l < 16; ++l) { float d = yr[l] - mu; var += d * d; }
    var *= (1.0f / 16);
    float rs = rsqrtf(var + EPS_);
    float denw = 1.0f / (rw + EPS_ * EPS_);
    for (int l = 0; l < 16; ++l) {
      float rec = (yr[l] - mu) * rs * A.dec_ln_g[l] + A.dec_ln_b[l];
      float rv  = (rec - rb) * denw * stdv + mean;
      int lg = tid * 16 + l;
      A.out_r[(b * L_ + lg) * C_ + c] = rv;
#if HAVE_TDM
      float xv = xs2[lg];
#else
      float xv = A.x[(b * L_ + lg) * C_ + c];
#endif
      float df = xv - rv;
      recp += df * df;
    }
  }
  red[tid] = recp;
  __syncthreads();
  for (int s = 128; s > 0; s >>= 1) {
    if (tid < s) red[tid] += red[tid + s];
    __syncthreads();
  }
  if (tid == 0) atomicAdd(A.ws_acc + 0, red[0]);
}

// ---------------- kernel 3: losses + perplexity ----------------

__global__ __launch_bounds__(512) void final_kernel(const float* ws_acc,
                                                    float* out_loss, float* out_rec,
                                                    float* out_perp) {
  __shared__ float red[512];
  int t = threadIdx.x;
  float cnt = ws_acc[2 + t];
  red[t] = cnt;
  __syncthreads();
  for (int s = 256; s > 0; s >>= 1) {
    if (t < s) red[t] += red[t + s];
    __syncthreads();
  }
  float total = red[0];
  __syncthreads();
  float p = cnt / (total + EPS_);
  red[t] = p * __logf(p + EPS_);
  __syncthreads();
  for (int s = 256; s > 0; s >>= 1) {
    if (t < s) red[t] += red[t + s];
    __syncthreads();
  }
  if (t == 0) {
    float perp = __expf(-red[0]);
    float rec  = ws_acc[0] * (1.0f / (float)(B_ * L_ * C_));
    float clu  = ws_acc[1] * (1.0f + BETA_) * (1.0f / (float)(B_ * C_ * P_ * 64));
    out_loss[0] = rec + CW_ * clu;
    out_rec[0]  = rec;
    out_perp[0] = perp;
  }
}

// ---------------- host launcher ----------------

extern "C" void kernel_launch(void* const* d_in, const int* in_sizes, int n_in,
                              void* d_out, int out_size, void* d_ws, size_t ws_size,
                              hipStream_t stream) {
  const float* x       = (const float*)d_in[0];
  const float* revin_w = (const float*)d_in[1];
  const float* revin_b = (const float*)d_in[2];
  const float* inp_W   = (const float*)d_in[3];
  const float* inp_b   = (const float*)d_in[4];
  const float* Wq      = (const float*)d_in[5];
  const float* bq      = (const float*)d_in[6];
  const float* Wk      = (const float*)d_in[7];
  const float* bk      = (const float*)d_in[8];
  const float* Wv      = (const float*)d_in[9];
  const float* bv      = (const float*)d_in[10];
  const float* Wo      = (const float*)d_in[11];
  const float* bo      = (const float*)d_in[12];
  const float* ln1_g   = (const float*)d_in[13];
  const float* ln1_b   = (const float*)d_in[14];
  const float* ffn_W1  = (const float*)d_in[15];
  const float* ffn_b1  = (const float*)d_in[16];
  const float* ffn_W2  = (const float*)d_in[17];
  const float* ffn_b2  = (const float*)d_in[18];
  const float* ln2_g   = (const float*)d_in[19];
  const float* ln2_b   = (const float*)d_in[20];
  const float* cb      = (const float*)d_in[21];
  const float* dec_W1  = (const float*)d_in[22];
  const float* dec_b1  = (const float*)d_in[23];
  const float* dec_W2  = (const float*)d_in[24];
  const float* dec_b2  = (const float*)d_in[25];
  const float* dec_Wr  = (const float*)d_in[26];
  const float* dec_br  = (const float*)d_in[27];
  const float* dec_ln_g = (const float*)d_in[28];
  const float* dec_ln_b = (const float*)d_in[29];

  float*    ws_acc  = (float*)d_ws;
  float*    ws_mean = (float*)((char*)d_ws + 4096);
  float*    ws_std  = (float*)((char*)d_ws + 8192);
  unsigned* wd      = (unsigned*)((char*)d_ws + 12288);
  float*    ws_e2   = (float*)((char*)d_ws + 223232);

  float* out = (float*)d_out;
  float* out_loss = out + 0;
  float* out_rec  = out + 1;
  float* out_zq   = out + 2;                       // [B,C,P,D] = 8388608
  float* out_r    = out + 2 + 8388608;             // [B,L,C]   = 2097152
  float* out_idx  = out + 2 + 8388608 + 2097152;   // [B,C,P]   = 131072
  float* out_perp = out + 2 + 8388608 + 2097152 + 131072;

  PrepArgs pa{inp_W, Wq, Wk, Wv, Wo, ffn_W1, ffn_W2, cb, dec_W1, dec_W2, dec_Wr,
              ws_acc, ws_e2, wd};
  prep_kernel<<<256, 256, 0, stream>>>(pa);

  revin_kernel<<<B_ * C_, 256, 0, stream>>>(x, ws_mean, ws_std);

  EncArgs ea;
  ea.x = x; ea.revin_w = revin_w; ea.revin_b = revin_b;
  ea.inp_b = inp_b; ea.bq = bq; ea.bk = bk; ea.bv = bv; ea.bo = bo;
  ea.ln1_g = ln1_g; ea.ln1_b = ln1_b; ea.ffn_b1 = ffn_b1; ea.ffn_b2 = ffn_b2;
  ea.ln2_g = ln2_g; ea.ln2_b = ln2_b;
  ea.codebook = cb; ea.dec_b1 = dec_b1; ea.dec_b2 = dec_b2; ea.dec_br = dec_br;
  ea.dec_ln_g = dec_ln_g; ea.dec_ln_b = dec_ln_b;
  ea.ws_mean = ws_mean; ea.ws_std = ws_std; ea.ws_e2 = ws_e2;
  ea.wd = wd; ea.ws_acc = ws_acc;
  ea.out_zq = out_zq; ea.out_r = out_r; ea.out_idx = out_idx;
  enc_kernel<<<B_ * C_, 256, 0, stream>>>(ea);

  final_kernel<<<1, 512, 0, stream>>>(ws_acc, out_loss, out_rec, out_perp);
}
